// SAGE_85177791414585
// MI455X (gfx1250) — matrix operand order repaired and checked by reading, not compiled, not run on hardware
//
#include <hip/hip_runtime.h>
#include <math.h>

typedef __attribute__((ext_vector_type(16))) _Float16 v16h;
typedef __attribute__((ext_vector_type(8)))  float    v8f;

#define NNODES 50000
#define NEDGES 640000

union AFrag { v16h v; _Float16 h[16]; };

__device__ __forceinline__ void cvt8(const float* __restrict__ p, float s, _Float16* d) {
    float4 a = *(const float4*)p;
    float4 b = *(const float4*)(p + 4);
    d[0] = (_Float16)(a.x * s); d[1] = (_Float16)(a.y * s);
    d[2] = (_Float16)(a.z * s); d[3] = (_Float16)(a.w * s);
    d[4] = (_Float16)(b.x * s); d[5] = (_Float16)(b.y * s);
    d[6] = (_Float16)(b.z * s); d[7] = (_Float16)(b.w * s);
}

// ---- degree kernels --------------------------------------------------------
__global__ void sage_deg(const int* __restrict__ dst, float* __restrict__ deg, int E) {
    int e = blockIdx.x * blockDim.x + threadIdx.x;
    if (e < E) atomicAdd(&deg[dst[e]], 1.0f);
}

__global__ void sage_rdeg(float* __restrict__ deg, int n) {
    int i = blockIdx.x * blockDim.x + threadIdx.x;
    if (i < n) deg[i] = 1.0f / fmaxf(deg[i], 1.0f);
}

// ---- edge scatter: agg[dst] += h[src], one wave per edge -------------------
__global__ __launch_bounds__(256) void sage_scatter(
    const float* __restrict__ h, const int* __restrict__ src,
    const int* __restrict__ dst, float* __restrict__ agg, int E)
{
    int e    = blockIdx.x * 8 + (threadIdx.x >> 5);
    int lane = threadIdx.x & 31;
    if (e >= E) return;
    int s = src[e], d = dst[e];
    float4 v = *(const float4*)(h + (size_t)s * 128 + lane * 4);
    float* p = agg + (size_t)d * 128 + lane * 4;
    atomicAdd(p + 0, v.x); atomicAdd(p + 1, v.y);
    atomicAdd(p + 2, v.z); atomicAdd(p + 3, v.w);
}

// ---- fused SAGE GEMM: out = relu?( [agg/deg | x] @ [Wl|Wr]^T + b ) ---------
// 256 threads = 8 waves; block computes 128 rows x NT*16 cols.
// K = 256 (128 mean-part against Wl, 128 root-part against Wr).
template<int NT, int NOUT, bool RELU>
__global__ __launch_bounds__(256) void sage_gemm(
    const float* __restrict__ agg, const float* __restrict__ x,
    const float* __restrict__ rdeg,
    const float* __restrict__ Wl, const float* __restrict__ Wr,
    const float* __restrict__ bias, float* __restrict__ out,
    int n_rows, int ldout)
{
    constexpr int KW = 272;  // 256 + 16 halves pad -> conflict-free strided rows
    __shared__ __align__(32) _Float16 sW[NT * 16][KW];

    const int tid = threadIdx.x;

    // Stage B = [Wl | Wr] rows (f32 -> f16), zero-pad n >= NOUT (layer 3).
    for (int idx = tid; idx < NT * 16 * 256; idx += 256) {
        int n = idx >> 8, k = idx & 255;
        float v = 0.0f;
        if (n < NOUT) v = (k < 128) ? Wl[n * 128 + k] : Wr[n * 128 + (k - 128)];
        sW[n][k] = (_Float16)v;
    }
    __syncthreads();

    const int wave = tid >> 5, lane = tid & 31;
    const int sel  = lane >> 4;       // half-wave select (ISA A/B/D layouts)
    const int l16  = lane & 15;
    const int mrow = blockIdx.x * 128 + wave * 16 + l16;
    const int r    = (mrow < n_rows) ? mrow : 0;
    const float rd = rdeg[r];

    // Full 16x256 A strip in registers, fp32->fp16 cvt inline.
    // A-frag k order used here (contiguous, same as B): lanes<16: K {k..k+15};
    // lanes>=16: shifted by 16.
    AFrag a[8];
#pragma unroll
    for (int ks = 0; ks < 4; ks++) {                 // mean part, K 0..127
        const float* base = agg + (size_t)r * 128 + ks * 32 + sel * 16;
        cvt8(base,      rd, &a[ks].h[0]);
        cvt8(base + 8, rd, &a[ks].h[8]);
    }
#pragma unroll
    for (int ks = 0; ks < 4; ks++) {                 // root part, K 128..255
        const float* base = x + (size_t)r * 128 + ks * 32 + sel * 16;
        cvt8(base,      1.0f, &a[4 + ks].h[0]);
        cvt8(base + 8, 1.0f, &a[4 + ks].h[8]);
    }

#pragma unroll
    for (int ct = 0; ct < NT; ct++) {
        v8f acc = {};
        // B-frag: column n = ct*16+l16 of B == row n of [Wl|Wr];
        // 16 contiguous halves at k = ks*32 + 16*sel.
        const _Float16* wrow = &sW[ct * 16 + l16][sel * 16];
#pragma unroll
        for (int ks = 0; ks < 8; ks++) {
            v16h b = *(const v16h*)(wrow + ks * 32);
            acc = __builtin_amdgcn_wmma_f32_16x16x32_f16(
                false, a[ks].v, false, b, (short)0, acc, false, false);
        }
        // D layout: lane holds n = l16, m = v + 8*sel within the 16-row tile.
        const int n = ct * 16 + l16;
        const float bv = (n < NOUT) ? bias[n] : 0.0f;
        const int mbase = blockIdx.x * 128 + wave * 16 + sel * 8;
#pragma unroll
        for (int v = 0; v < 8; v++) {
            int m = mbase + v;
            if (m < n_rows && n < NOUT) {
                float rr = acc[v] + bv;
                if (RELU) rr = fmaxf(rr, 0.0f);
                out[(size_t)m * ldout + n] = rr;
            }
        }
    }
}

// ---- row-wise log_softmax over 40 cols, one wave per row -------------------
__global__ __launch_bounds__(256) void sage_lsm(
    const float* __restrict__ in, float* __restrict__ out, int n)
{
    int row  = blockIdx.x * 8 + (threadIdx.x >> 5);
    int lane = threadIdx.x & 31;
    if (row >= n) return;
    const float* p = in + (size_t)row * 40;
    float v0 = p[lane];
    float v1 = (lane < 8) ? p[lane + 32] : -INFINITY;
    float mx = fmaxf(v0, v1);
#pragma unroll
    for (int o = 16; o; o >>= 1) mx = fmaxf(mx, __shfl_xor(mx, o, 32));
    float s = expf(v0 - mx) + ((lane < 8) ? expf(v1 - mx) : 0.0f);
#pragma unroll
    for (int o = 16; o; o >>= 1) s += __shfl_xor(s, o, 32);
    float lse = mx + logf(s);
    out[(size_t)row * 40 + lane] = v0 - lse;
    if (lane < 8) out[(size_t)row * 40 + lane + 32] = v1 - lse;
}

// ---------------------------------------------------------------------------
extern "C" void kernel_launch(void* const* d_in, const int* in_sizes, int n_in,
                              void* d_out, int out_size, void* d_ws, size_t ws_size,
                              hipStream_t stream)
{
    const float* x   = (const float*)d_in[0];
    const int*   ei  = (const int*)d_in[1];
    const float* W1l = (const float*)d_in[2];
    const float* b1  = (const float*)d_in[3];
    const float* W1r = (const float*)d_in[4];
    const float* W2l = (const float*)d_in[5];
    const float* b2  = (const float*)d_in[6];
    const float* W2r = (const float*)d_in[7];
    const float* W3l = (const float*)d_in[8];
    const float* b3  = (const float*)d_in[9];
    const float* W3r = (const float*)d_in[10];
    float* out = (float*)d_out;

    const int N = NNODES, E = NEDGES;
    const int* src = ei;
    const int* dst = ei + E;

    float* ws   = (float*)d_ws;
    float* agg  = ws;                         // [N,128]
    float* h1   = agg + (size_t)N * 128;      // [N,128]
    float* h2   = h1  + (size_t)N * 128;      // [N,128]
    float* h3   = h2  + (size_t)N * 128;      // [N,40]
    float* rdeg = h3  + (size_t)N * 40;       // [N]

    // degree (same graph for all layers): count, then invert
    hipMemsetAsync(rdeg, 0, (size_t)N * sizeof(float), stream);
    sage_deg<<<(E + 255) / 256, 256, 0, stream>>>(dst, rdeg, E);
    sage_rdeg<<<(N + 255) / 256, 256, 0, stream>>>(rdeg, N);

    const int gScatter = (E + 7) / 8;
    const int gGemm    = (N + 127) / 128;

    // layer 1
    hipMemsetAsync(agg, 0, (size_t)N * 128 * sizeof(float), stream);
    sage_scatter<<<gScatter, 256, 0, stream>>>(x, src, dst, agg, E);
    sage_gemm<8, 128, true><<<gGemm, 256, 0, stream>>>(agg, x, rdeg, W1l, W1r, b1, h1, N, 128);

    // layer 2
    hipMemsetAsync(agg, 0, (size_t)N * 128 * sizeof(float), stream);
    sage_scatter<<<gScatter, 256, 0, stream>>>(h1, src, dst, agg, E);
    sage_gemm<8, 128, true><<<gGemm, 256, 0, stream>>>(agg, h1, rdeg, W2l, W2r, b2, h2, N, 128);

    // layer 3 (OUT=40, padded to 48 cols, no relu)
    hipMemsetAsync(agg, 0, (size_t)N * 128 * sizeof(float), stream);
    sage_scatter<<<gScatter, 256, 0, stream>>>(h2, src, dst, agg, E);
    sage_gemm<3, 40, false><<<gGemm, 256, 0, stream>>>(agg, h2, rdeg, W3l, W3r, b3, h3, N, 40);

    // log_softmax
    sage_lsm<<<(N + 7) / 8, 256, 0, stream>>>(h3, out, N);
}
